// GCNChannel_63084479644063
// MI455X (gfx1250) — matrix operand order repaired and checked by reading, not compiled, not run on hardware
//
#include <hip/hip_runtime.h>
#include <hip/hip_bf16.h>

#define BATCH 4
#define CH    64
#define HIN   128
#define WIN   128
#define H1    64
#define W1    64
#define NPIX  4096   // H1*W1

typedef __attribute__((ext_vector_type(16))) __bf16        v16bf;
typedef __attribute__((ext_vector_type(8)))  unsigned int  v8u;
typedef __attribute__((ext_vector_type(8)))  float         v8f;

// ---------- bf16 helpers (round-to-nearest-even) ----------
__device__ __forceinline__ unsigned short f2bf(float f) {
  unsigned int u = __builtin_bit_cast(unsigned int, f);
  u += 0x7FFFu + ((u >> 16) & 1u);
  return (unsigned short)(u >> 16);
}
__device__ __forceinline__ float bf2f(unsigned short h) {
  unsigned int u = ((unsigned int)h) << 16;
  return __builtin_bit_cast(float, u);
}
__device__ __forceinline__ v8f zero8() {
  v8f z;
#pragma unroll
  for (int i = 0; i < 8; ++i) z[i] = 0.f;
  return z;
}

// ---------- WMMA fragment builders (CDNA5 16x16x32 bf16 layouts) ----------
// A matrix 16x32 (MxK), src row-major, K contiguous.
// lane L: row = row0 + L%16 ; VGPR j holds K pair base (j<4 ? 2j : 2j+8) + 8*(L/16)
// -> two 16B contiguous groups (bytes 0..15 and 32..47): compiler emits 2x b128.
__device__ __forceinline__ v16bf load_frag_a(const unsigned short* src, int ld,
                                             int row0, int k0) {
  int lane = threadIdx.x & 31;
  const unsigned short* p =
      src + (size_t)(row0 + (lane & 15)) * ld + k0 + ((lane >> 4) << 3);
  v8u f;
#pragma unroll
  for (int j = 0; j < 8; ++j) {
    int kb = (j < 4) ? (2 * j) : (2 * j + 8);
    f[j] = *(const unsigned int*)(p + kb);
  }
  return __builtin_bit_cast(v16bf, f);
}
// B matrix 32x16 (KxN), src stored col-major-as-rows: src[col*ld + K], K contiguous.
// lane L: col = col0 + L%16 ; element e: K = (e<8 ? e : e+8) + 8*(L/16)  -> two 16B groups: 2x b128.
__device__ __forceinline__ v16bf load_frag_b(const unsigned short* src, int ld,
                                             int col0, int k0) {
  int lane = threadIdx.x & 31;
  const unsigned short* p =
      src + (size_t)(col0 + (lane & 15)) * ld + k0 + ((lane >> 4) << 3);
  v8u f;
#pragma unroll
  for (int j = 0; j < 8; ++j) f[j] = *(const unsigned int*)(p + ((j < 4) ? (2 * j) : (2 * j + 8)));
  return __builtin_bit_cast(v16bf, f);
}
__device__ __forceinline__ v8f wmma_bf16(v16bf a, v16bf b, v8f c) {
  return __builtin_amdgcn_wmma_f32_16x16x32_bf16(false, a, false, b,
                                                 (short)0, c, false, false);
}

// ---------- K1: conv3x3 s2 p1 + bias + BN + ReLU -> xa (b, n, c) bf16 ----------
__global__ void __launch_bounds__(256)
conv_bn_relu_kernel(const float* __restrict__ x, const float* __restrict__ w,
                    const float* __restrict__ bias, const float* __restrict__ gamma,
                    const float* __restrict__ beta, const float* __restrict__ mean,
                    const float* __restrict__ var, unsigned short* __restrict__ xa) {
  __shared__ float wsh[CH * 9];
  int tid = threadIdx.x;
  int co = blockIdx.y, b = blockIdx.z;
  for (int i = tid; i < CH * 9; i += 256) wsh[i] = w[(size_t)co * CH * 9 + i];
  __syncthreads();
  int p  = blockIdx.x * 256 + tid;
  int oh = p >> 6, ow = p & 63;
  int ih0 = oh * 2 - 1, iw0 = ow * 2 - 1;
  float acc = 0.f;
  for (int ci = 0; ci < CH; ++ci) {
    const float* xp = x + ((size_t)b * CH + ci) * HIN * WIN;
    const float* wp = wsh + ci * 9;
#pragma unroll
    for (int kh = 0; kh < 3; ++kh) {
      int ih = ih0 + kh;
      if ((unsigned)ih >= HIN) continue;
#pragma unroll
      for (int kw = 0; kw < 3; ++kw) {
        int iw = iw0 + kw;
        if ((unsigned)iw >= WIN) continue;
        acc = fmaf(xp[ih * WIN + iw], wp[kh * 3 + kw], acc);
      }
    }
  }
  float s = gamma[co] * rsqrtf(var[co] + 1e-5f);
  float v = (acc + bias[co] - mean[co]) * s + beta[co];
  v = fmaxf(v, 0.f);
  xa[((size_t)b * NPIX + p) * CH + co] = f2bf(v);
}

// ---------- K2: Sobel |gx|,|gy| on the (N x c) image, zero-padded ----------
__global__ void __launch_bounds__(256)
sobel_kernel(const unsigned short* __restrict__ xa, unsigned short* __restrict__ sx,
             unsigned short* __restrict__ sy) {
  size_t idx = (size_t)blockIdx.x * 256 + threadIdx.x;   // over b*N*c
  int c = (int)(idx & 63);
  int n = (int)((idx >> 6) & (NPIX - 1));
  int b = (int)(idx >> 18);
  float v[3][3];
#pragma unroll
  for (int dn = -1; dn <= 1; ++dn)
#pragma unroll
    for (int dc = -1; dc <= 1; ++dc) {
      int nn = n + dn, cc = c + dc;
      float t = 0.f;
      if ((unsigned)nn < NPIX && (unsigned)cc < CH)
        t = bf2f(xa[((size_t)b * NPIX + nn) * CH + cc]);
      v[dn + 1][dc + 1] = t;
    }
  float gx = (v[0][2] - v[0][0]) + 2.f * (v[1][2] - v[1][0]) + (v[2][2] - v[2][0]);
  float gy = (v[2][0] - v[0][0]) + 2.f * (v[2][1] - v[0][1]) + (v[2][2] - v[0][2]);
  sx[idx] = f2bf(fabsf(gx));
  sy[idx] = f2bf(fabsf(gy));
}

// ---------- K3: A-tile GEMM (sx·sy^T, K=64) fused with batch-softmax + I ----------
// One block computes the 64x64 (n,m) tile for ALL 4 batches (softmax couples them).
__global__ void __launch_bounds__(256)
attention_softmax_kernel(const unsigned short* __restrict__ sx,
                         const unsigned short* __restrict__ sy,
                         unsigned short* __restrict__ P) {
  int m0 = blockIdx.x * 64, n0 = blockIdx.y * 64;
  int wave = threadIdx.x >> 5, lane = threadIdx.x & 31;
  int wr = wave & 3;            // n-subtile 0..3
  int wc = (wave >> 2) * 2;     // m-subtiles {wc, wc+1}
  v8f acc[BATCH][2];
#pragma unroll
  for (int bb = 0; bb < BATCH; ++bb) { acc[bb][0] = zero8(); acc[bb][1] = zero8(); }
#pragma unroll
  for (int bb = 0; bb < BATCH; ++bb) {
    const unsigned short* sxb = sx + (size_t)bb * NPIX * CH;
    const unsigned short* syb = sy + (size_t)bb * NPIX * CH;
#pragma unroll
    for (int ks = 0; ks < CH; ks += 32) {
      v16bf fa = load_frag_a(sxb, CH, n0 + wr * 16, ks);
      v16bf f0 = load_frag_b(syb, CH, m0 + wc * 16, ks);
      v16bf f1 = load_frag_b(syb, CH, m0 + (wc + 1) * 16, ks);
      acc[bb][0] = wmma_bf16(fa, f0, acc[bb][0]);
      acc[bb][1] = wmma_bf16(fa, f1, acc[bb][1]);
    }
  }
  int nrow_base = n0 + wr * 16 + ((lane >> 4) << 3);
#pragma unroll
  for (int s = 0; s < 2; ++s) {
    int mcol = m0 + (wc + s) * 16 + (lane & 15);
#pragma unroll
    for (int r = 0; r < 8; ++r) {
      int nrow = nrow_base + r;
      float l0 = acc[0][s][r], l1 = acc[1][s][r];
      float l2 = acc[2][s][r], l3 = acc[3][s][r];
      float mx = fmaxf(fmaxf(l0, l1), fmaxf(l2, l3));
      float e0 = __expf(l0 - mx), e1 = __expf(l1 - mx);
      float e2 = __expf(l2 - mx), e3 = __expf(l3 - mx);
      float inv = 1.f / (e0 + e1 + e2 + e3);
      float idv = (nrow == mcol) ? 1.f : 0.f;
      size_t base = (size_t)nrow * NPIX + mcol;
      P[(size_t)0 * NPIX * NPIX + base] = f2bf(e0 * inv + idv);
      P[(size_t)1 * NPIX * NPIX + base] = f2bf(e1 * inv + idv);
      P[(size_t)2 * NPIX * NPIX + base] = f2bf(e2 * inv + idv);
      P[(size_t)3 * NPIX * NPIX + base] = f2bf(e3 * inv + idv);
    }
  }
}

// ---------- K4: row sums of P (includes +I) -> dinv = rsqrt(sum) ----------
__global__ void __launch_bounds__(256)
rowsum_dinv_kernel(const unsigned short* __restrict__ P, float* __restrict__ dinv) {
  __shared__ float red[256];
  size_t row = blockIdx.x;                 // b*NPIX + n
  const unsigned short* pr = P + row * NPIX;
  float s = 0.f;
  for (int i = threadIdx.x; i < NPIX; i += 256) s += bf2f(pr[i]);
  red[threadIdx.x] = s;
  __syncthreads();
  for (int off = 128; off > 0; off >>= 1) {
    if (threadIdx.x < (unsigned)off) red[threadIdx.x] += red[threadIdx.x + off];
    __syncthreads();
  }
  if (threadIdx.x == 0) dinv[row] = rsqrtf(red[0]);
}

// ---------- K5: H = dinv_m * (X @ W^T), written TRANSPOSED: Ht[b][o][m] bf16 ----------
// Wave owns 16 rows x all 64 cols (4 accumulators); block = 128 rows.
__global__ void __launch_bounds__(256)
xw_scale_kernel(const unsigned short* __restrict__ X, const float* __restrict__ W,
                const float* __restrict__ dinv, unsigned short* __restrict__ Ht) {
  __shared__ unsigned short wsh[CH * CH];
  int b = blockIdx.y, tid = threadIdx.x;
  for (int i = tid; i < CH * CH; i += 256) wsh[i] = f2bf(W[i]);
  __syncthreads();
  int wave = tid >> 5, lane = tid & 31;
  const unsigned short* Xb = X + (size_t)b * NPIX * CH;
  int mrow = blockIdx.x * 128 + wave * 16;
  v8f acc[4];
#pragma unroll
  for (int s = 0; s < 4; ++s) acc[s] = zero8();
#pragma unroll
  for (int ks = 0; ks < CH; ks += 32) {
    v16bf fa = load_frag_a(Xb, CH, mrow, ks);
#pragma unroll
    for (int s = 0; s < 4; ++s) {
      v16bf fb = load_frag_b(wsh, CH, s * 16, ks);
      acc[s] = wmma_bf16(fa, fb, acc[s]);
    }
  }
  int mrow_base = mrow + ((lane >> 4) << 3);
#pragma unroll
  for (int s = 0; s < 4; ++s) {
    int ocol = s * 16 + (lane & 15);
#pragma unroll
    for (int r = 0; r < 8; ++r) {
      int m = mrow_base + r;
      float v = acc[s][r] * dinv[(size_t)b * NPIX + m];
      Ht[((size_t)b * CH + ocol) * NPIX + m] = f2bf(v);
    }
  }
}

// ---------- K6: Y = act(dinv_n * (P @ H)) ; 4096x4096x64 GEMM per batch ----------
// Wave owns 16 rows x all 64 cols: 2.5 b128 loads per wmma; A (P-stream) loaded once.
__global__ void __launch_bounds__(256)
propagate_kernel(const unsigned short* __restrict__ P, const unsigned short* __restrict__ Ht,
                 const float* __restrict__ dinv, unsigned short* __restrict__ Xout,
                 float* __restrict__ Yout, int do_relu) {
  int b = blockIdx.y;
  int wave = threadIdx.x >> 5, lane = threadIdx.x & 31;
  const unsigned short* Pb = P + (size_t)b * NPIX * NPIX;
  const unsigned short* Hb = Ht + (size_t)b * CH * NPIX;
  int arow = blockIdx.x * 128 + wave * 16;
  v8f acc[4];
#pragma unroll
  for (int s = 0; s < 4; ++s) acc[s] = zero8();
  // per-lane base of this wave's A-row stream, for software prefetch ahead
  const unsigned short* pfbase =
      Pb + (size_t)(arow + (lane & 15)) * NPIX + ((lane >> 4) << 3);
#pragma unroll 2
  for (int ks = 0; ks < NPIX; ks += 32) {
    if (ks + 256 < NPIX)                       // warm L2/WGP$ ~8 chunks (512B) ahead
      __builtin_prefetch(pfbase + ks + 256, 0, 3);
    v16bf fa = load_frag_a(Pb, NPIX, arow, ks);
#pragma unroll
    for (int s = 0; s < 4; ++s) {
      v16bf fb = load_frag_b(Hb, NPIX, s * 16, ks);
      acc[s] = wmma_bf16(fa, fb, acc[s]);
    }
  }
  int nrow_base = arow + ((lane >> 4) << 3);
#pragma unroll
  for (int s = 0; s < 4; ++s) {
    int ocol = s * 16 + (lane & 15);
#pragma unroll
    for (int r = 0; r < 8; ++r) {
      int n = nrow_base + r;
      float v = acc[s][r] * dinv[(size_t)b * NPIX + n];
      if (do_relu) {
        v = fmaxf(v, 0.f);
        Xout[((size_t)b * NPIX + n) * CH + ocol] = f2bf(v);
      } else {
        Yout[((size_t)b * CH + ocol) * NPIX + n] = v;   // 'bon' layout
      }
    }
  }
}

// ---------- K7: bilinear align-corners 64 -> 128 ----------
__global__ void __launch_bounds__(256)
upsample_kernel(const float* __restrict__ Yin, float* __restrict__ out) {
  size_t idx = (size_t)blockIdx.x * 256 + threadIdx.x;   // (b,c,oh,ow)
  int ow = (int)(idx & 127);
  int oh = (int)((idx >> 7) & 127);
  size_t bc = idx >> 14;
  const float sc = 63.0f / 127.0f;
  float fy = oh * sc, fx = ow * sc;
  int y0 = (int)fy, x0 = (int)fx;
  int y1 = min(y0 + 1, 63), x1 = min(x0 + 1, 63);
  float wy = fy - (float)y0, wx = fx - (float)x0;
  const float* src = Yin + bc * NPIX;
  float t = src[y0 * 64 + x0] * (1.f - wx) + src[y0 * 64 + x1] * wx;
  float bm = src[y1 * 64 + x0] * (1.f - wx) + src[y1 * 64 + x1] * wx;
  out[idx] = t * (1.f - wy) + bm * wy;
}

extern "C" void kernel_launch(void* const* d_in, const int* in_sizes, int n_in,
                              void* d_out, int out_size, void* d_ws, size_t ws_size,
                              hipStream_t stream) {
  const float* x        = (const float*)d_in[0];
  const float* conv_w   = (const float*)d_in[1];
  const float* conv_b   = (const float*)d_in[2];
  const float* bn_gamma = (const float*)d_in[3];
  const float* bn_beta  = (const float*)d_in[4];
  const float* bn_mean  = (const float*)d_in[5];
  const float* bn_var   = (const float*)d_in[6];
  const float* fc1_w    = (const float*)d_in[7];
  const float* fc2_w    = (const float*)d_in[8];
  const float* fc3_w    = (const float*)d_in[9];
  float* out = (float*)d_out;

  // workspace carve-up (~142 MB total, P last at 128 MB)
  char* ws = (char*)d_ws;
  const size_t szBF = (size_t)BATCH * NPIX * CH * sizeof(unsigned short);  // 2 MB
  unsigned short* xa  = (unsigned short*)ws; ws += szBF;
  unsigned short* xb  = (unsigned short*)ws; ws += szBF;
  unsigned short* ht  = (unsigned short*)ws; ws += szBF;
  unsigned short* sxb = (unsigned short*)ws; ws += szBF;
  unsigned short* syb = (unsigned short*)ws; ws += szBF;
  float* dinv = (float*)ws; ws += (size_t)BATCH * NPIX * sizeof(float);
  float* yout = (float*)ws; ws += (size_t)BATCH * CH * NPIX * sizeof(float);
  unsigned short* P = (unsigned short*)ws;   // BATCH*NPIX*NPIX bf16 = 128 MB

  conv_bn_relu_kernel<<<dim3(NPIX / 256, CH, BATCH), 256, 0, stream>>>(
      x, conv_w, conv_b, bn_gamma, bn_beta, bn_mean, bn_var, xa);
  sobel_kernel<<<(BATCH * NPIX * CH) / 256, 256, 0, stream>>>(xa, sxb, syb);
  attention_softmax_kernel<<<dim3(NPIX / 64, NPIX / 64), 256, 0, stream>>>(sxb, syb, P);
  rowsum_dinv_kernel<<<BATCH * NPIX, 256, 0, stream>>>(P, dinv);

  // stage 1: xa -> xb (relu)
  xw_scale_kernel<<<dim3(NPIX / 128, BATCH), 256, 0, stream>>>(xa, fc1_w, dinv, ht);
  propagate_kernel<<<dim3(NPIX / 128, BATCH), 256, 0, stream>>>(P, ht, dinv, xb, nullptr, 1);
  // stage 2: xb -> xa (relu)
  xw_scale_kernel<<<dim3(NPIX / 128, BATCH), 256, 0, stream>>>(xb, fc2_w, dinv, ht);
  propagate_kernel<<<dim3(NPIX / 128, BATCH), 256, 0, stream>>>(P, ht, dinv, xa, nullptr, 1);
  // stage 3: xa -> yout (no act, 'bon' transposed f32)
  xw_scale_kernel<<<dim3(NPIX / 128, BATCH), 256, 0, stream>>>(xa, fc3_w, dinv, ht);
  propagate_kernel<<<dim3(NPIX / 128, BATCH), 256, 0, stream>>>(P, ht, dinv, nullptr, yout, 0);

  upsample_kernel<<<(BATCH * CH * HIN * WIN) / 256, 256, 0, stream>>>(yout, out);
}
